// CustomAttention_36077725286944
// MI455X (gfx1250) — matrix operand order repaired and checked by reading, not compiled, not run on hardware
//
#include <hip/hip_runtime.h>
#include <math.h>

// ---------------------------------------------------------------------------
// Shapes (fixed by the reference):  B=4, S=2048, D_MODEL=1024, scale=1/8
// ---------------------------------------------------------------------------
#define BATCH   4
#define SEQ     2048
#define DMODEL  1024
#define SCALE   0.125f
#define CH      64                     // attention columns processed per chunk

typedef __attribute__((ext_vector_type(16))) __bf16 v16bf;
typedef __attribute__((ext_vector_type(8)))  float  v8f;

union BF16Tile { v16bf v; unsigned u[8]; uint4 q[2]; };

__device__ __forceinline__ unsigned short f2bf(float f) {
    unsigned x = __builtin_bit_cast(unsigned, f);
    x = x + 0x7fffu + ((x >> 16) & 1u);           // round-to-nearest-even
    return (unsigned short)(x >> 16);
}
__device__ __forceinline__ unsigned pack2(float lo, float hi) {
    return (unsigned)f2bf(lo) | ((unsigned)f2bf(hi) << 16);
}
__device__ __forceinline__ float rmax16(float x) {
    #pragma unroll
    for (int m = 1; m < 16; m <<= 1) x = fmaxf(x, __shfl_xor(x, m, 32));
    return x;
}
__device__ __forceinline__ float rsum16(float x) {
    #pragma unroll
    for (int m = 1; m < 16; m <<= 1) x += __shfl_xor(x, m, 32);
    return x;
}

// ---------------------------------------------------------------------------
// Prologue 1: row-major f32 -> bf16 pairs (Q with scale folded, K, W_out).
// ---------------------------------------------------------------------------
__global__ __launch_bounds__(256) void cvt_bf16_kernel(
        const float* __restrict__ src, unsigned* __restrict__ dst, float scale) {
    size_t i = ((size_t)blockIdx.x * 256 + threadIdx.x) * 4;
    float4 f = *reinterpret_cast<const float4*>(src + i);
    dst[(i >> 1) + 0] = pack2(f.x * scale, f.y * scale);
    dst[(i >> 1) + 1] = pack2(f.z * scale, f.w * scale);
}

// ---------------------------------------------------------------------------
// Prologue 2: V [b][k][col] f32 -> Vt [b][col][k] bf16 (WMMA B-operand order).
// ---------------------------------------------------------------------------
__global__ __launch_bounds__(256) void transpose_v_kernel(
        const float* __restrict__ V, unsigned* __restrict__ Vt) {
    __shared__ unsigned short t[64][66];

    const int tid = threadIdx.x;
    const int k0  = blockIdx.x * 64;
    const int c0  = blockIdx.y * 64;
    const int b   = blockIdx.z;

    const int c  = tid & 63;
    const int r0 = tid >> 6;
    #pragma unroll
    for (int i = 0; i < 16; ++i) {
        int row = r0 + i * 4;
        float f = V[((size_t)(b * SEQ + k0 + row)) * DMODEL + c0 + c];
        t[c][row] = f2bf(f);
    }
    __syncthreads();

    const int kp = tid & 31;
    const int cb = tid >> 5;
    #pragma unroll
    for (int p = 0; p < 8; ++p) {
        int col = cb + p * 8;
        unsigned u = *reinterpret_cast<const unsigned*>(&t[col][2 * kp]);
        Vt[((size_t)(b * DMODEL + c0 + col)) * (SEQ / 2) + (k0 >> 1) + kp] = u;
    }
}

// ---------------------------------------------------------------------------
// Kernel 1: flash attention.  Grid: BATCH*SEQ/16 workgroups, 256 threads
// (8 waves).  64-column chunks: 4 score tiles merged in LDS (ds_add_f32),
// one softmax/rescale round per 32 WMMAs per wave.  Q tile staged via
// global_load_async_to_lds_b128 (ASYNCcnt).
// ---------------------------------------------------------------------------
__global__ __launch_bounds__(256) void attn_fa_kernel(
        const unsigned* __restrict__ Qbf,   // bf16 pairs, pre-scaled, [b*S][D]
        const unsigned* __restrict__ Kbf,   // bf16 pairs, row-major [b*S][D]
        const unsigned* __restrict__ Vt,    // bf16 pairs, [b][col][k]
        unsigned short* __restrict__ O) {   // bf16 out [b*S][D]

    __shared__ __align__(16) unsigned lQ[16 * (DMODEL / 2)];  // 32 KB
    __shared__ float    lS[16 * CH];                          // 4 KB
    __shared__ unsigned lPu[16 * (CH / 2)];                   // 2 KB

    const int tid  = threadIdx.x;
    const int wave = tid >> 5;
    const int lane = tid & 31;
    const int half = lane >> 4;
    const int cc   = lane & 15;

    const int b  = blockIdx.x / (SEQ / 16);
    const int qt = blockIdx.x % (SEQ / 16);
    const size_t qoff = (size_t)(b * SEQ + qt * 16) * DMODEL;

    // ---- async-stage the Q tile (bf16) straight into LDS ----
    {
        const uint4* qsrc = reinterpret_cast<const uint4*>(Qbf) + (qoff >> 3);
        unsigned lbase = (unsigned)(uintptr_t)(void*)lQ;
        #pragma unroll
        for (int i = 0; i < 8; ++i) {
            int idx = i * 256 + tid;                    // 2048 x 16B = 32 KB
            asm volatile("global_load_async_to_lds_b128 %0, %1, off"
                         :: "v"(lbase + idx * 16), "v"(qsrc + idx) : "memory");
        }
        asm volatile("s_wait_asynccnt 0x0" ::: "memory");
    }
    __syncthreads();

    const uint4* lq4 = reinterpret_cast<const uint4*>(lQ);
    const uint4* lp4 = reinterpret_cast<const uint4*>(lPu);

    float mrow[8], lrow[8];
    v8f   acc[8];
    #pragma unroll
    for (int r = 0; r < 8; ++r) { mrow[r] = -INFINITY; lrow[r] = 0.f; }
    #pragma unroll
    for (int nt = 0; nt < 8; ++nt) acc[nt] = (v8f){};

    for (int chunk = 0; chunk < SEQ / CH; ++chunk) {
        const int kbase = chunk * CH;

        reinterpret_cast<float4*>(lS)[tid] = (float4){0.f, 0.f, 0.f, 0.f};
        __syncthreads();

        // ---- QK^T partials over this wave's D-range [wave*128, +128) ----
        #pragma unroll 2
        for (int t = 0; t < 4; ++t) {                   // 4 x 16-col score tiles
            v8f s = (v8f){};
            const size_t krow = (size_t)(b * SEQ + kbase + t * 16 + cc) * (DMODEL / 2);
            #pragma unroll
            for (int kk = 0; kk < 4; ++kk) {
                const int dBase = wave * 128 + kk * 32;
                BF16Tile A, Bm;
                A.q[0] = lq4[cc * 128 + (dBase >> 3) + half];
                A.q[1] = lq4[cc * 128 + (dBase >> 3) + half + 2];
                const uint4* kb = reinterpret_cast<const uint4*>(
                                      Kbf + krow + (dBase >> 1) + 4 * half);
                Bm.q[0] = kb[0];
                Bm.q[1] = kb[2];
                s = __builtin_amdgcn_wmma_f32_16x16x32_bf16(
                        false, A.v, false, Bm.v, (short)0, s, false, false);
            }
            #pragma unroll
            for (int r = 0; r < 8; ++r) {
                int m = r + 8 * half;
                atomicAdd(&lS[m * CH + t * 16 + cc], s[r]);
            }
        }
        __syncthreads();

        // ---- online softmax over the 64 merged columns ----
        float p[4][8];
        #pragma unroll
        for (int r = 0; r < 8; ++r) {
            int   m = r + 8 * half;
            float s0 = lS[m * CH + cc];
            float s1 = lS[m * CH + 16 + cc];
            float s2 = lS[m * CH + 32 + cc];
            float s3 = lS[m * CH + 48 + cc];
            float mx = fmaxf(fmaxf(s0, s1), fmaxf(s2, s3));
            float mn = fmaxf(mrow[r], rmax16(mx));
            float al = __expf(mrow[r] - mn);
            p[0][r] = __expf(s0 - mn);
            p[1][r] = __expf(s1 - mn);
            p[2][r] = __expf(s2 - mn);
            p[3][r] = __expf(s3 - mn);
            lrow[r] = lrow[r] * al +
                      rsum16((p[0][r] + p[1][r]) + (p[2][r] + p[3][r]));
            mrow[r] = mn;
            #pragma unroll
            for (int nt = 0; nt < 8; ++nt) acc[nt][r] *= al;
        }
        if (wave == 0) {                                // publish P (16 x 64, bf16)
            unsigned short* lPh = reinterpret_cast<unsigned short*>(lPu);
            #pragma unroll
            for (int r = 0; r < 8; ++r) {
                int m = r + 8 * half;
                #pragma unroll
                for (int t = 0; t < 4; ++t)
                    lPh[m * CH + t * 16 + cc] = f2bf(p[t][r]);
            }
        }
        __syncthreads();

        // ---- PV: acc[nt] += P(16x64) @ V(64x16), 8 column tiles per wave ----
        BF16Tile Ap0, Ap1;
        Ap0.q[0] = lp4[cc * 8 + half];
        Ap0.q[1] = lp4[cc * 8 + half + 2];
        Ap1.q[0] = lp4[cc * 8 + 4 + half];
        Ap1.q[1] = lp4[cc * 8 + 4 + half + 2];
        #pragma unroll
        for (int nt = 0; nt < 8; ++nt) {
            const int col = wave * 128 + nt * 16 + cc;
            const uint4* vb = reinterpret_cast<const uint4*>(
                Vt + (size_t)(b * DMODEL + col) * (SEQ / 2) + (kbase >> 1)) + half;
            BF16Tile Bv0, Bv1;
            Bv0.q[0] = vb[0];
            Bv0.q[1] = vb[2];
            Bv1.q[0] = vb[4];
            Bv1.q[1] = vb[6];
            acc[nt] = __builtin_amdgcn_wmma_f32_16x16x32_bf16(
                          false, Ap0.v, false, Bv0.v, (short)0, acc[nt], false, false);
            acc[nt] = __builtin_amdgcn_wmma_f32_16x16x32_bf16(
                          false, Ap1.v, false, Bv1.v, (short)0, acc[nt], false, false);
        }
        __syncthreads();
    }

    // ---- normalize and store attention output (bf16) into workspace ----
    #pragma unroll
    for (int r = 0; r < 8; ++r) {
        float inv = 1.0f / lrow[r];
        int   m   = r + 8 * half;
        size_t row = (size_t)(b * SEQ + qt * 16 + m);
        #pragma unroll
        for (int nt = 0; nt < 8; ++nt) {
            int col = wave * 128 + nt * 16 + cc;
            O[row * DMODEL + col] = f2bf(acc[nt][r] * inv);
        }
    }
}

// ---------------------------------------------------------------------------
// Kernel 2: output projection  y = O @ W^T, all-bf16 operands, f32 result.
// ---------------------------------------------------------------------------
__global__ __launch_bounds__(32) void out_proj_kernel(
        const unsigned* __restrict__ Ou,
        const unsigned* __restrict__ Wbf,
        float* __restrict__ Y) {

    const int lane = threadIdx.x;
    const int half = lane >> 4;
    const int cc   = lane & 15;
    const int rt   = blockIdx.x;
    const int ct   = blockIdx.y;

    v8f acc[4];
    #pragma unroll
    for (int nt = 0; nt < 4; ++nt) acc[nt] = (v8f){};

    const size_t arow = (size_t)(rt * 16 + cc) * (DMODEL / 2);

    for (int dc = 0; dc < DMODEL / 32; ++dc) {
        const int dBase = dc * 32;
        BF16Tile A;
        const uint4* ap = reinterpret_cast<const uint4*>(
                              Ou + arow + (dBase >> 1) + 4 * half);
        A.q[0] = ap[0];
        A.q[1] = ap[2];
        #pragma unroll
        for (int nt = 0; nt < 4; ++nt) {
            const size_t erow = (size_t)(ct * 64 + nt * 16 + cc) * (DMODEL / 2);
            const uint4* bp = reinterpret_cast<const uint4*>(
                                  Wbf + erow + (dBase >> 1) + 4 * half);
            BF16Tile Bw;
            Bw.q[0] = bp[0];
            Bw.q[1] = bp[2];
            acc[nt] = __builtin_amdgcn_wmma_f32_16x16x32_bf16(
                          false, A.v, false, Bw.v, (short)0, acc[nt], false, false);
        }
    }

    #pragma unroll
    for (int nt = 0; nt < 4; ++nt) {
        #pragma unroll
        for (int r = 0; r < 8; ++r) {
            size_t row = (size_t)(rt * 16 + r + 8 * half);
            int    e   = ct * 64 + nt * 16 + cc;
            Y[row * DMODEL + e] = acc[nt][r];
        }
    }
}

// ---------------------------------------------------------------------------
extern "C" void kernel_launch(void* const* d_in, const int* in_sizes, int n_in,
                              void* d_out, int out_size, void* d_ws, size_t ws_size,
                              hipStream_t stream) {
    const float* q = (const float*)d_in[0];
    const float* k = (const float*)d_in[1];
    const float* v = (const float*)d_in[2];
    const float* w = (const float*)d_in[3];

    const size_t NKV = (size_t)BATCH * SEQ * DMODEL;                 // 8M elems
    unsigned short* base = (unsigned short*)d_ws;
    unsigned short* O    = base;                                      // 16 MB
    unsigned*       Kbf  = (unsigned*)(base + NKV);                   // 16 MB
    unsigned*       Vt   = (unsigned*)(base + 2 * NKV);               // 16 MB
    unsigned*       Wbf  = (unsigned*)(base + 3 * NKV);               //  2 MB
    unsigned*       Qbf  = (unsigned*)(base + 3 * NKV + DMODEL * DMODEL); // 16 MB

    cvt_bf16_kernel<<<dim3(NKV / (256 * 4)), dim3(256), 0, stream>>>(q, Qbf, SCALE);
    cvt_bf16_kernel<<<dim3(NKV / (256 * 4)), dim3(256), 0, stream>>>(k, Kbf, 1.0f);
    cvt_bf16_kernel<<<dim3((DMODEL * DMODEL) / (256 * 4)), dim3(256), 0, stream>>>(
        w, Wbf, 1.0f);
    transpose_v_kernel<<<dim3(SEQ / 64, DMODEL / 64, BATCH), dim3(256), 0, stream>>>(
        v, Vt);

    attn_fa_kernel<<<dim3(BATCH * (SEQ / 16)), dim3(256), 0, stream>>>(
        Qbf, Kbf, Vt, O);
    out_proj_kernel<<<dim3((BATCH * SEQ) / 16, DMODEL / 64), dim3(32), 0, stream>>>(
        (const unsigned*)O, Wbf, (float*)d_out);
}
